// LM_19396072308891
// MI455X (gfx1250) — matrix operand order repaired and checked by reading, not compiled, not run on hardware
//
#include <hip/hip_runtime.h>

// ---------------------------------------------------------------------------
// Problem constants (from reference)
// ---------------------------------------------------------------------------
#define S_  5
#define B_  32
#define T_  25
#define TM_ 24          // T-1 timesteps actually processed
#define V_  32000
#define E_  512
#define H_  1024
#define NBLK_GRU 32     // persistent GRU kernel block count (must all be resident)

typedef __attribute__((ext_vector_type(16))) __bf16        v16bf;
typedef __attribute__((ext_vector_type(8)))  float         v8f;
typedef __attribute__((ext_vector_type(4)))  unsigned int  v4u;
typedef __attribute__((ext_vector_type(8)))  unsigned int  v8u;

// f32 -> bf16 (round to nearest even), raw ushort storage
__device__ __forceinline__ unsigned short f2bf(float f) {
    unsigned int u = __float_as_uint(f);
    u += 0x7FFFu + ((u >> 16) & 1u);
    return (unsigned short)(u >> 16);
}

__device__ __forceinline__ float sigmoidf_(float x) {
    return 1.0f / (1.0f + __expf(-x));
}

// ---------------------------------------------------------------------------
// WMMA fragment loaders (wave32).
// A (16xK, 16-bit): lane m<16 holds row m, K-chunks [k..k+7] and [k+16..k+23];
//                   lane m+16 holds row m, K-chunks [k+8..k+15] and [k+24..k+31].
// B (Kx16, 16-bit): lane n<16 holds col n, K [k..k+7] and [k+16..k+23]; lane n+16 col n, K [k+8..k+15] and [k+24..k+31].
// W matrices are stored (N,K) row-major == B column-major: col n contiguous in K.
// ---------------------------------------------------------------------------
__device__ __forceinline__ v16bf load_a_frag(const unsigned short* __restrict__ base,
                                             int ld, int row0, int k, int lane) {
    int r  = row0 + (lane & 15);
    int kc = k + ((lane >> 4) << 3);          // 0 or 8
    const unsigned short* p = base + (size_t)r * ld + kc;
    v8u w;
    w.lo = *(const v4u*)(p);                  // k-chunk +0
    w.hi = *(const v4u*)(p + 16);             // k-chunk +16
    return __builtin_bit_cast(v16bf, w);
}

__device__ __forceinline__ v16bf load_b_frag(const unsigned short* __restrict__ base,
                                             int ld, int col0, int k, int lane) {
    int c  = col0 + (lane & 15);
    int kc = k + ((lane >> 4) << 3);          // 0 or 8
    const unsigned short* p = base + (size_t)c * ld + kc;
    v8u w;
    w.lo = *(const v4u*)(p);
    w.hi = *(const v4u*)(p + 16);
    return __builtin_bit_cast(v16bf, w);
}

__device__ __forceinline__ v8f wmma_bf16(v16bf a, v16bf b, v8f c) {
    return __builtin_amdgcn_wmma_f32_16x16x32_bf16(false, a, false, b, (short)0, c,
                                                   false, false);
}

// ---------------------------------------------------------------------------
// Grid-wide barrier for the persistent GRU kernel (all NBLK_GRU blocks resident).
// Release on arrive (agent-scope RMW), acquire spin, s_sleep while waiting.
// `target` is the cumulative arrival count to wait for (monotonic epochs).
// ---------------------------------------------------------------------------
__device__ __forceinline__ void grid_barrier(unsigned int* cnt, unsigned int target) {
    __threadfence();              // make this thread's global stores visible (agent)
    __syncthreads();
    if (threadIdx.x == 0) {
        __hip_atomic_fetch_add(cnt, 1u, __ATOMIC_ACQ_REL, __HIP_MEMORY_SCOPE_AGENT);
        while (__hip_atomic_load(cnt, __ATOMIC_ACQUIRE, __HIP_MEMORY_SCOPE_AGENT) < target) {
            __builtin_amdgcn_s_sleep(2);
        }
    }
    __syncthreads();
}

// ---------------------------------------------------------------------------
// Utility kernels
// ---------------------------------------------------------------------------
__global__ void cvt_f32_bf16_kernel(const float* __restrict__ in,
                                    unsigned short* __restrict__ out, int n) {
    int i = blockIdx.x * blockDim.x + threadIdx.x;
    if (i < n) out[i] = f2bf(in[i]);
}

// Gather embeddings for tokens[:, :, :TM_] into x_bf laid out [s][t][b][E]
__global__ void embed_gather_kernel(const int* __restrict__ ids,
                                    const float* __restrict__ embed,
                                    unsigned short* __restrict__ x_bf) {
    int row = blockIdx.x;                       // (s*TM + t)*B + b
    int s = row / (TM_ * B_);
    int rem = row % (TM_ * B_);
    int t = rem / B_;
    int b = rem % B_;
    int tok = ids[(s * B_ + b) * T_ + t];
    const float* src = embed + (size_t)tok * E_;
    unsigned short* dst = x_bf + (size_t)row * E_;
    for (int i = threadIdx.x; i < E_; i += blockDim.x)
        dst[i] = f2bf(src[i]);
}

__global__ void zero_state_kernel(float* __restrict__ cf,
                                  unsigned short* __restrict__ cb,
                                  unsigned int* __restrict__ bar_cnt, int n) {
    int i = blockIdx.x * blockDim.x + threadIdx.x;
    if (i < n) { cf[i] = 0.0f; cb[i] = 0; }
    if (i == 0) *bar_cnt = 0u;
}

// carry[l][b][:] = hs_all[clip(len-2,0,TM-1)][l][b][:]
__global__ void carry_select_kernel(const int* __restrict__ seq_lens_s,
                                    const float* __restrict__ hs_all,
                                    float* __restrict__ carry_f,
                                    unsigned short* __restrict__ carry_b) {
    int b = blockIdx.x;
    int l = blockIdx.y;
    int idx = seq_lens_s[b] - 2;
    idx = idx < 0 ? 0 : (idx > TM_ - 1 ? TM_ - 1 : idx);
    const float* src = hs_all + (((size_t)idx * 2 + l) * B_ + b) * H_;
    float* dst = carry_f + ((size_t)l * B_ + b) * H_;
    unsigned short* dbf = carry_b + ((size_t)l * B_ + b) * H_;
    for (int i = threadIdx.x; i < H_; i += blockDim.x) {
        float v = src[i];
        dst[i] = v;
        dbf[i] = f2bf(v);
    }
}

// ---------------------------------------------------------------------------
// One GRU layer, one wave's (16 batch x 16 hidden) tile:
// gi = x@Wih^T, gh = h@Whh^T via WMMA (f32 acc), then gates + state update.
// ---------------------------------------------------------------------------
__device__ __forceinline__ void gru_tile(
    const unsigned short* __restrict__ xa,   // [B][Kx] bf16 input activations
    int Kx,
    const unsigned short* __restrict__ Wih,  // [3H][Kx] bf16
    const unsigned short* __restrict__ Whh,  // [3H][H]  bf16
    const float* __restrict__ bih,
    const float* __restrict__ bhh,
    const unsigned short* __restrict__ h_in_bf,  // [B][H]
    const float* __restrict__ h_in_f32,          // [B][H]
    float* __restrict__ h_out_f32,               // [B][H]
    unsigned short* __restrict__ h_out_bf,       // [B][H]
    int mb, int jb, int lane)
{
    v8f gi_r = {}, gi_z = {}, gi_n = {};
    v8f gh_r = {}, gh_z = {}, gh_n = {};

    for (int k = 0; k < Kx; k += 32) {
        v16bf a  = load_a_frag(xa, Kx, mb, k, lane);
        v16bf br = load_b_frag(Wih, Kx, jb,          k, lane);
        v16bf bz = load_b_frag(Wih, Kx, H_ + jb,     k, lane);
        v16bf bn = load_b_frag(Wih, Kx, 2 * H_ + jb, k, lane);
        gi_r = wmma_bf16(a, br, gi_r);
        gi_z = wmma_bf16(a, bz, gi_z);
        gi_n = wmma_bf16(a, bn, gi_n);
    }
    for (int k = 0; k < H_; k += 32) {
        v16bf a  = load_a_frag(h_in_bf, H_, mb, k, lane);
        v16bf br = load_b_frag(Whh, H_, jb,          k, lane);
        v16bf bz = load_b_frag(Whh, H_, H_ + jb,     k, lane);
        v16bf bn = load_b_frag(Whh, H_, 2 * H_ + jb, k, lane);
        gh_r = wmma_bf16(a, br, gh_r);
        gh_z = wmma_bf16(a, bz, gh_z);
        gh_n = wmma_bf16(a, bn, gh_n);
    }

    const int col = jb + (lane & 15);
    const float bir = bih[col], biz = bih[H_ + col], bin = bih[2 * H_ + col];
    const float bhr = bhh[col], bhz = bhh[H_ + col], bhn = bhh[2 * H_ + col];
    const int rowbase = mb + ((lane >> 4) << 3);
#pragma unroll
    for (int v = 0; v < 8; ++v) {
        const int row = rowbase + v;
        float r = sigmoidf_(gi_r[v] + bir + gh_r[v] + bhr);
        float z = sigmoidf_(gi_z[v] + biz + gh_z[v] + bhz);
        float n = tanhf(gi_n[v] + bin + r * (gh_n[v] + bhn));
        float hp = h_in_f32[(size_t)row * H_ + col];
        float hn = (1.0f - z) * n + z * hp;
        h_out_f32[(size_t)row * H_ + col] = hn;
        h_out_bf[(size_t)row * H_ + col] = f2bf(hn);
    }
}

// ---------------------------------------------------------------------------
// Persistent per-sentence GRU scan: 32 blocks x 4 waves = 128 waves
// (= 2 batch-tiles x 64 hidden-tiles). Iterates t=0..23 internally with one
// grid barrier per timestep (between layer0 and layer1; arrival at barrier(t)
// also proves all blocks finished layer1(t-1) by program order).
// ---------------------------------------------------------------------------
__global__ __launch_bounds__(128) void gru_sentence_kernel(
    const unsigned short* __restrict__ x_s,   // [TM][B][E] this sentence
    const unsigned short* __restrict__ Wih0, const unsigned short* __restrict__ Whh0,
    const float* __restrict__ bih0, const float* __restrict__ bhh0,
    const unsigned short* __restrict__ Wih1, const unsigned short* __restrict__ Whh1,
    const float* __restrict__ bih1, const float* __restrict__ bhh1,
    const float* __restrict__ carryf, const unsigned short* __restrict__ carryb,
    float* __restrict__ hs_all,               // [TM][2][B][H] f32 history
    unsigned short* __restrict__ hbfA,
    unsigned short* __restrict__ hbfB,        // layer0 bf16 ping-pong
    unsigned short* __restrict__ hs_top_s,    // [TM][B][H] layer1 bf16 out
    unsigned int* bar_cnt, unsigned int epoch0)
{
    const int lane = threadIdx.x & 31;
    const int wid  = blockIdx.x * 4 + (threadIdx.x >> 5);
    const int mb   = (wid & 1) << 4;
    const int jb   = (wid >> 1) << 4;
    unsigned short* hbf[2] = { hbfA, hbfB };

    for (int t = 0; t < TM_; ++t) {
        const unsigned short* x = x_s + (size_t)t * B_ * E_;

        const unsigned short* h0inbf = (t == 0) ? carryb : hbf[(t - 1) & 1];
        const float* h0inf = (t == 0) ? carryf
                                      : hs_all + ((size_t)(t - 1) * 2) * B_ * H_;
        float* h0outf = hs_all + ((size_t)t * 2) * B_ * H_;
        unsigned short* h0outbf = hbf[t & 1];
        gru_tile(x, E_, Wih0, Whh0, bih0, bhh0,
                 h0inbf, h0inf, h0outf, h0outbf, mb, jb, lane);

        grid_barrier(bar_cnt, (epoch0 + (unsigned)t + 1u) * (unsigned)NBLK_GRU);

        const unsigned short* h1inbf =
            (t == 0) ? (carryb + (size_t)B_ * H_)
                     : (hs_top_s + (size_t)(t - 1) * B_ * H_);
        const float* h1inf = (t == 0) ? (carryf + (size_t)B_ * H_)
                                      : hs_all + ((size_t)(t - 1) * 2 + 1) * B_ * H_;
        float* h1outf = hs_all + ((size_t)t * 2 + 1) * B_ * H_;
        unsigned short* h1outbf = hs_top_s + (size_t)t * B_ * H_;
        gru_tile(h0outbf, H_, Wih1, Whh1, bih1, bhh1,
                 h1inbf, h1inf, h1outf, h1outbf, mb, jb, lane);
    }
}

// ---------------------------------------------------------------------------
// Logits GEMM: out[s][b][t][v] = hs_top[(s*TM+t)*B + b][:] . W_out[v][:]
// Block = 4 waves, block tile 32(M) x 256(N); wave tile 32 x 64 (8 accums,
// B fragments reused across both M-tiles). The block's 32-row A panel is
// staged into LDS with global_load_async_to_lds_b128 (ASYNCcnt), K split in
// two 512-wide stages; padded rows skew LDS banks. Grid: (125, 120).
// ---------------------------------------------------------------------------
#define KSTG  512
#define SAPAD (KSTG + 8)     // elements per staged row (16B pad -> bank skew)

__global__ __launch_bounds__(128) void logits_kernel(
    const unsigned short* __restrict__ A,   // hs_top [3840][H] bf16
    const unsigned short* __restrict__ Bw,  // W_out  [V][H]    bf16
    float* __restrict__ out)                // [S][B][TM][V] f32
{
    __shared__ unsigned short smemA[32 * SAPAD];

    const int lane = threadIdx.x & 31;
    const int w    = threadIdx.x >> 5;
    const int nb   = blockIdx.x * 256 + w * 64;   // wave's N base
    const int mr0  = blockIdx.y * 32;             // block's M base

    const unsigned smem_base =
        (unsigned)(unsigned long long)(const void*)&smemA[0];

    v8f acc[2][4];
#pragma unroll
    for (int mt = 0; mt < 2; ++mt)
#pragma unroll
        for (int j = 0; j < 4; ++j) acc[mt][j] = (v8f){};

    for (int k0 = 0; k0 < H_; k0 += KSTG) {
        __syncthreads();   // previous stage fully consumed before overwrite
        // Async-stage A[mr0..mr0+32)[k0..k0+KSTG) -> LDS (16B chunks per lane)
        for (int c = threadIdx.x; c < 32 * (KSTG / 8); c += 128) {
            int row = c >> 6;                 // KSTG/8 = 64 chunks per row
            int ke  = (c & 63) << 3;          // element offset within stage
            unsigned ldsb = smem_base + (unsigned)((row * SAPAD + ke) * 2);
            unsigned long long ga = (unsigned long long)(const void*)
                (A + (size_t)(mr0 + row) * H_ + k0 + ke);
            asm volatile("global_load_async_to_lds_b128 %0, %1, off"
                         :: "v"(ldsb), "v"(ga) : "memory");
        }
        asm volatile("s_wait_asynccnt 0x0" ::: "memory");
        __syncthreads();

        for (int k = 0; k < KSTG; k += 32) {
            // A fragments from LDS (ds_load_b128 pairs)
            v16bf a0, a1;
            {
                int r  = lane & 15;
                int kc = k + ((lane >> 4) << 3);
                const unsigned short* p = &smemA[r * SAPAD + kc];
                v8u t0; t0.lo = *(const v4u*)p; t0.hi = *(const v4u*)(p + 16);
                a0 = __builtin_bit_cast(v16bf, t0);
                const unsigned short* q = &smemA[(r + 16) * SAPAD + kc];
                v8u t1; t1.lo = *(const v4u*)q; t1.hi = *(const v4u*)(q + 16);
                a1 = __builtin_bit_cast(v16bf, t1);
            }
            const int kg = k0 + k;
#pragma unroll
            for (int j = 0; j < 4; ++j) {
                v16bf b = load_b_frag(Bw, H_, nb + j * 16, kg, lane);
                acc[0][j] = wmma_bf16(a0, b, acc[0][j]);
                acc[1][j] = wmma_bf16(a1, b, acc[1][j]);
            }
        }
    }

    const int colb   = lane & 15;
    const int rowoff = (lane >> 4) << 3;
#pragma unroll
    for (int mt = 0; mt < 2; ++mt) {
#pragma unroll
        for (int v = 0; v < 8; ++v) {
            const int r   = mr0 + mt * 16 + rowoff + v;   // (s*TM+t)*B + b
            const int s   = r / (TM_ * B_);
            const int rem = r % (TM_ * B_);
            const int t   = rem >> 5;
            const int b   = rem & 31;
            const size_t ob = (((size_t)(s * B_ + b)) * TM_ + t) * (size_t)V_;
#pragma unroll
            for (int j = 0; j < 4; ++j)
                out[ob + nb + j * 16 + colb] = acc[mt][j][v];
        }
    }
}

// ---------------------------------------------------------------------------
// Host launcher
// ---------------------------------------------------------------------------
extern "C" void kernel_launch(void* const* d_in, const int* in_sizes, int n_in,
                              void* d_out, int out_size, void* d_ws, size_t ws_size,
                              hipStream_t stream) {
    (void)in_sizes; (void)n_in; (void)out_size; (void)ws_size;

    const int*   input_ids = (const int*)  d_in[0];
    const int*   seq_lens  = (const int*)  d_in[1];
    const float* embed     = (const float*)d_in[2];
    const float* W_ih0     = (const float*)d_in[3];
    const float* W_hh0     = (const float*)d_in[4];
    const float* b_ih0     = (const float*)d_in[5];
    const float* b_hh0     = (const float*)d_in[6];
    const float* W_ih1     = (const float*)d_in[7];
    const float* W_hh1     = (const float*)d_in[8];
    const float* b_ih1     = (const float*)d_in[9];
    const float* b_hh1     = (const float*)d_in[10];
    const float* W_out     = (const float*)d_in[11];
    float* out = (float*)d_out;

    // Workspace carving (256B aligned)
    char* ws = (char*)d_ws;
    auto alloc = [&](size_t bytes) -> void* {
        void* p = (void*)ws;
        ws += (bytes + 255) & ~(size_t)255;
        return p;
    };
    unsigned short* Wih0b  = (unsigned short*)alloc((size_t)3 * H_ * E_ * 2);
    unsigned short* Whh0b  = (unsigned short*)alloc((size_t)3 * H_ * H_ * 2);
    unsigned short* Wih1b  = (unsigned short*)alloc((size_t)3 * H_ * H_ * 2);
    unsigned short* Whh1b  = (unsigned short*)alloc((size_t)3 * H_ * H_ * 2);
    unsigned short* Woutb  = (unsigned short*)alloc((size_t)V_ * H_ * 2);
    unsigned short* x_bf   = (unsigned short*)alloc((size_t)S_ * TM_ * B_ * E_ * 2);
    unsigned short* hs_top = (unsigned short*)alloc((size_t)S_ * TM_ * B_ * H_ * 2);
    float*          hs_all = (float*)alloc((size_t)TM_ * 2 * B_ * H_ * 4);
    float*          carryf = (float*)alloc((size_t)2 * B_ * H_ * 4);
    unsigned short* carryb = (unsigned short*)alloc((size_t)2 * B_ * H_ * 2);
    unsigned short* hbf0_0 = (unsigned short*)alloc((size_t)B_ * H_ * 2);
    unsigned short* hbf0_1 = (unsigned short*)alloc((size_t)B_ * H_ * 2);
    unsigned int*   barcnt = (unsigned int*)  alloc(256);

    // Weight / embedding conversion (every call; deterministic)
    {
        int n;
        n = 3 * H_ * E_;
        cvt_f32_bf16_kernel<<<(n + 255) / 256, 256, 0, stream>>>(W_ih0, Wih0b, n);
        n = 3 * H_ * H_;
        cvt_f32_bf16_kernel<<<(n + 255) / 256, 256, 0, stream>>>(W_hh0, Whh0b, n);
        cvt_f32_bf16_kernel<<<(n + 255) / 256, 256, 0, stream>>>(W_ih1, Wih1b, n);
        cvt_f32_bf16_kernel<<<(n + 255) / 256, 256, 0, stream>>>(W_hh1, Whh1b, n);
        n = V_ * H_;
        cvt_f32_bf16_kernel<<<(n + 255) / 256, 256, 0, stream>>>(W_out, Woutb, n);
    }
    embed_gather_kernel<<<S_ * TM_ * B_, 128, 0, stream>>>(input_ids, embed, x_bf);
    {
        int n = 2 * B_ * H_;
        zero_state_kernel<<<(n + 255) / 256, 256, 0, stream>>>(carryf, carryb,
                                                               barcnt, n);
    }

    // Sequential 2-layer GRU scan: one persistent kernel per sentence
    for (int s = 0; s < S_; ++s) {
        gru_sentence_kernel<<<NBLK_GRU, 128, 0, stream>>>(
            x_bf + (size_t)s * TM_ * B_ * E_,
            Wih0b, Whh0b, b_ih0, b_hh0,
            Wih1b, Whh1b, b_ih1, b_hh1,
            carryf, carryb, hs_all, hbf0_0, hbf0_1,
            hs_top + (size_t)s * TM_ * B_ * H_,
            barcnt, (unsigned)(s * TM_));
        carry_select_kernel<<<dim3(B_, 2), 256, 0, stream>>>(
            seq_lens + s * B_, hs_all, carryf, carryb);
    }

    // Output projection: 3840 x 32000 x 1024 bf16 WMMA GEMM
    logits_kernel<<<dim3(V_ / 256, (S_ * TM_ * B_) / 32), 128, 0, stream>>>(
        hs_top, Woutb, out);
}
